// GPT2Model_4286377361410
// MI455X (gfx1250) — matrix operand order repaired and checked by reading, not compiled, not run on hardware
//
#include <hip/hip_runtime.h>
#include <hip/hip_bf16.h>

// ---------------------------------------------------------------------------
// GPT2 (no-softmax attention) forward for MI455X / gfx1250.
// Linear attention reassociation: (Q K^T) V * s  ==  Q (K^T V) * s.
// GEMMs on v_wmma_f32_16x16x32_bf16 with async-to-LDS double-buffered tiles.
// ---------------------------------------------------------------------------

typedef __bf16 bf16_t;
typedef __attribute__((ext_vector_type(16))) __bf16 v16bf;
typedef __attribute__((ext_vector_type(8)))  __bf16 v8bf;
typedef __attribute__((ext_vector_type(8)))  float  v8f;
typedef __attribute__((ext_vector_type(4)))  int    v4i;

typedef __attribute__((address_space(1))) v4i* gptr4;  // global int4*
typedef __attribute__((address_space(3))) v4i* lptr4;  // LDS int4*

#define NLAYER 12
#define BB 4
#define TT 2048
#define CC 1024
#define HH 16
#define DD 64
#define MTOK (BB * TT) /* 8192 tokens */

#define BK 32      /* K-step per WMMA stage */
#define LDSROW 40  /* padded LDS row stride (elements): 80B -> 20-bank stride */

#if __has_builtin(__builtin_amdgcn_global_load_async_to_lds_b128) && \
    __has_builtin(__builtin_amdgcn_s_wait_asynccnt)
#define USE_ASYNC_LDS 1
#else
#define USE_ASYNC_LDS 0
#endif

// ----------------------------- helpers -------------------------------------

__device__ __forceinline__ bf16_t f2bf(float f) {
  union { float f; unsigned u; } x; x.f = f;
  unsigned r = (x.u + 0x7FFFu + ((x.u >> 16) & 1u)) >> 16;  // round-nearest-even
  unsigned short s = (unsigned short)r;
  bf16_t out; __builtin_memcpy(&out, &s, 2);
  return out;
}

__device__ __forceinline__ v16bf combine16(v8bf lo, v8bf hi) {
  v16bf out;
#pragma unroll
  for (int i = 0; i < 8; ++i) { out[i] = lo[i]; out[i + 8] = hi[i]; }
  return out;
}

// A fragment (16x32 bf16) straight from row-major global memory.
__device__ __forceinline__ v16bf load_fragA(const bf16_t* __restrict__ A,
                                            int lda, int m0, int k0) {
  const int lane = threadIdx.x & 31;
  const int r  = lane & 15;
  const int kc = (lane >> 4) * 8;
  const bf16_t* p = A + (size_t)(m0 + r) * lda + (k0 + kc);
  return combine16(*(const v8bf*)(p), *(const v8bf*)(p + 16));
}

// B fragment (32x16 bf16) straight from K-major global memory (rows of W).
__device__ __forceinline__ v16bf load_fragB(const bf16_t* __restrict__ W,
                                            int ldw, int n0, int k0) {
  const int lane = threadIdx.x & 31;
  const int n  = lane & 15;
  const int kc = (lane >> 4) * 8;
  const bf16_t* p = W + (size_t)(n0 + n) * ldw + (k0 + kc);
  return combine16(*(const v8bf*)(p), *(const v8bf*)(p + 16));
}

// Fragments from an LDS-staged [128 x BK] tile with padded rows.
__device__ __forceinline__ v16bf fragA_lds(const bf16_t* buf, int m_local) {
  const int lane = threadIdx.x & 31;
  const int r  = lane & 15;
  const int kc = (lane >> 4) * 8;
  const bf16_t* p = buf + (m_local + r) * LDSROW + kc;
  return combine16(*(const v8bf*)(p), *(const v8bf*)(p + 16));
}
__device__ __forceinline__ v16bf fragB_lds(const bf16_t* buf, int n_local) {
  const int lane = threadIdx.x & 31;
  const int n  = lane & 15;
  const int kc = (lane >> 4) * 8;
  const bf16_t* p = buf + (n_local + n) * LDSROW + kc;
  return combine16(*(const v8bf*)(p), *(const v8bf*)(p + 16));
}

__device__ __forceinline__ v8f wmma_bf16(v16bf a, v16bf b, v8f c) {
  return __builtin_amdgcn_wmma_f32_16x16x32_bf16(
      /*neg_a=*/false, a, /*neg_b=*/false, b,
      /*c_mod=*/(short)0, c, /*reuse_a=*/false, /*reuse_b=*/false);
}

// Stage a 128-row x 64B (BK bf16) tile into LDS: 512 x 16B chunks across
// 256 threads (2 chunks each), via the gfx1250 async-to-LDS path.
__device__ __forceinline__ void stage_tile(bf16_t* __restrict__ lds,
                                           const bf16_t* __restrict__ g,
                                           int ldg, int row0, int k0) {
  const int tid = threadIdx.x;
#pragma unroll
  for (int c = 0; c < 2; ++c) {
    const int chunk = tid + c * 256;
    const int row = chunk >> 2;
    const int col = chunk & 3;  // 16B units within the 64B row
    const bf16_t* src = g + (size_t)(row0 + row) * ldg + k0 + col * 8;
    bf16_t* dst = lds + row * LDSROW + col * 8;
#if USE_ASYNC_LDS
    __builtin_amdgcn_global_load_async_to_lds_b128((gptr4)src, (lptr4)dst, 0, 0);
#else
    *(v8bf*)dst = *(const v8bf*)src;  // fallback: load + ds_store
#endif
  }
}

__device__ __forceinline__ void stage_fence() {
#if USE_ASYNC_LDS
  __builtin_amdgcn_s_wait_asynccnt(0);
#endif
  __syncthreads();
}

// ----------------------------- kernels -------------------------------------

__global__ void k_init(const float* __restrict__ x0, float* __restrict__ xf,
                       bf16_t* __restrict__ xb, int n) {
  for (int i = blockIdx.x * blockDim.x + threadIdx.x; i < n;
       i += gridDim.x * blockDim.x) {
    float v = x0[i];
    xf[i] = v;
    xb[i] = f2bf(v);
  }
}

__global__ void k_cvt(const float* __restrict__ src, bf16_t* __restrict__ dst,
                      int n) {
  for (int i = blockIdx.x * blockDim.x + threadIdx.x; i < n;
       i += gridDim.x * blockDim.x)
    dst[i] = f2bf(src[i]);
}

// qkv = X @ Wqkv^T + b ; scatter Q -> [B,H,T,D], K,V -> transposed [B,H,D,T].
// Block tile 128x128 (8 waves, wave tile 32x64), async-LDS double buffered.
__global__ __launch_bounds__(256)
void k_gemm_qkv(const bf16_t* __restrict__ X,   // [8192,1024]
                const bf16_t* __restrict__ W,   // [3072,1024] bf16
                const float* __restrict__ bias, // [3072]
                bf16_t* __restrict__ q,         // [B,H,T,D]
                bf16_t* __restrict__ kT,        // [B,H,D,T]
                bf16_t* __restrict__ vT) {      // [B,H,D,T]
  __shared__ __attribute__((aligned(32))) bf16_t sA[2][128 * LDSROW];
  __shared__ __attribute__((aligned(32))) bf16_t sB[2][128 * LDSROW];

  const int wave = threadIdx.x >> 5;
  const int wm = wave & 3, wn = wave >> 2;
  const int mblk = blockIdx.y * 128, nblk = blockIdx.x * 128;
  const int mloc = wm * 32, nloc = wn * 64;

  stage_tile(sA[0], X, CC, mblk, 0);
  stage_tile(sB[0], W, CC, nblk, 0);
  stage_fence();

  v8f acc[2][4] = {};
  int buf = 0;
  for (int k0 = 0; k0 < CC; k0 += BK, buf ^= 1) {
    if (k0 + BK < CC) {
      stage_tile(sA[buf ^ 1], X, CC, mblk, k0 + BK);
      stage_tile(sB[buf ^ 1], W, CC, nblk, k0 + BK);
      __builtin_prefetch(W + (size_t)(nblk + (threadIdx.x >> 1)) * CC + k0 + 2 * BK, 0, 3);
    }
    v16bf a0 = fragA_lds(sA[buf], mloc);
    v16bf a1 = fragA_lds(sA[buf], mloc + 16);
#pragma unroll
    for (int j = 0; j < 4; ++j) {
      v16bf b = fragB_lds(sB[buf], nloc + j * 16);
      acc[0][j] = wmma_bf16(a0, b, acc[0][j]);
      acc[1][j] = wmma_bf16(a1, b, acc[1][j]);
    }
    stage_fence();
  }

  const int lane = threadIdx.x & 31;
  const int ncol = lane & 15;
  const int mr0 = (lane >> 4) * 8;
#pragma unroll
  for (int i = 0; i < 2; ++i) {
#pragma unroll
    for (int j = 0; j < 4; ++j) {
      const int ng = nblk + nloc + j * 16 + ncol;
      const float bv = bias[ng];
#pragma unroll
      for (int r = 0; r < 8; ++r) {
        const int mg = mblk + mloc + i * 16 + mr0 + r;
        const bf16_t o = f2bf(acc[i][j][r] + bv);
        const int b = mg >> 11, t = mg & (TT - 1);
        if (ng < CC) {
          const int h = ng >> 6, d = ng & 63;
          q[(((size_t)(b * HH + h)) * TT + t) * DD + d] = o;
        } else if (ng < 2 * CC) {
          const int nn = ng - CC, h = nn >> 6, d = nn & 63;
          kT[(((size_t)(b * HH + h)) * DD + d) * TT + t] = o;
        } else {
          const int nn = ng - 2 * CC, h = nn >> 6, d = nn & 63;
          vT[(((size_t)(b * HH + h)) * DD + d) * TT + t] = o;
        }
      }
    }
  }
}

// S' = (V^T K) * scale  per (b,h): M=N=64, K=2048. Output S^T [64,64] bf16.
__global__ __launch_bounds__(256)
void k_gemm_svt(const bf16_t* __restrict__ vT, const bf16_t* __restrict__ kT,
                bf16_t* __restrict__ sT, float scale) {
  const int bh = blockIdx.x;
  const bf16_t* A  = vT + (size_t)bh * DD * TT;
  const bf16_t* Bm = kT + (size_t)bh * DD * TT;
  const int wave = threadIdx.x >> 5;
  const int m0 = (wave >> 1) * 16;
  const int nt = (wave & 1) * 2;

  v8f acc[2] = {};
  for (int k0 = 0; k0 < TT; k0 += BK) {
    v16bf a = load_fragA(A, TT, m0, k0);
#pragma unroll
    for (int j = 0; j < 2; ++j) {
      v16bf b = load_fragB(Bm, TT, (nt + j) * 16, k0);
      acc[j] = wmma_bf16(a, b, acc[j]);
    }
  }
  const int lane = threadIdx.x & 31;
  const int ncol = lane & 15;
  const int mr0 = (lane >> 4) * 8;
#pragma unroll
  for (int j = 0; j < 2; ++j)
#pragma unroll
    for (int r = 0; r < 8; ++r)
      sT[(size_t)bh * DD * DD + (m0 + mr0 + r) * DD + (nt + j) * 16 + ncol] =
          f2bf(acc[j][r] * scale);
}

// Y = Q @ S  per (b,h): M=2048, N=64, K=64. Output merged to [B,T,C] bf16.
__global__ __launch_bounds__(256)
void k_gemm_y(const bf16_t* __restrict__ q, const bf16_t* __restrict__ sT,
              bf16_t* __restrict__ y) {
  const int bh = blockIdx.x;
  const int b = bh >> 4, h = bh & 15;
  const bf16_t* A  = q  + (size_t)bh * TT * DD;
  const bf16_t* Bm = sT + (size_t)bh * DD * DD;
  const int wave = threadIdx.x >> 5;
  const int m0 = blockIdx.y * 128 + wave * 16;

  v8f acc[4] = {};
#pragma unroll
  for (int k0 = 0; k0 < DD; k0 += BK) {
    v16bf a = load_fragA(A, DD, m0, k0);
#pragma unroll
    for (int j = 0; j < 4; ++j) {
      v16bf bfr = load_fragB(Bm, DD, j * 16, k0);
      acc[j] = wmma_bf16(a, bfr, acc[j]);
    }
  }
  const int lane = threadIdx.x & 31;
  const int ncol = lane & 15;
  const int mr0 = (lane >> 4) * 8;
#pragma unroll
  for (int j = 0; j < 4; ++j)
#pragma unroll
    for (int r = 0; r < 8; ++r) {
      const int t = m0 + mr0 + r;
      const int d = j * 16 + ncol;
      y[((size_t)(b * TT + t)) * CC + h * DD + d] = f2bf(acc[j][r]);
    }
}

// Xnew = Y @ Wproj^T + bias + Xold (fp32 state in d_out); bf16 copy for next
// layer. Same async-LDS double-buffered structure as k_gemm_qkv.
__global__ __launch_bounds__(256)
void k_gemm_proj(const bf16_t* __restrict__ Y,  // [8192,1024]
                 const bf16_t* __restrict__ W,  // [1024,1024]
                 const float* __restrict__ bias,
                 float* __restrict__ X,         // fp32 residual, in/out
                 bf16_t* __restrict__ Xb) {
  __shared__ __attribute__((aligned(32))) bf16_t sA[2][128 * LDSROW];
  __shared__ __attribute__((aligned(32))) bf16_t sB[2][128 * LDSROW];

  const int wave = threadIdx.x >> 5;
  const int wm = wave & 3, wn = wave >> 2;
  const int mblk = blockIdx.y * 128, nblk = blockIdx.x * 128;
  const int mloc = wm * 32, nloc = wn * 64;

  stage_tile(sA[0], Y, CC, mblk, 0);
  stage_tile(sB[0], W, CC, nblk, 0);
  stage_fence();

  v8f acc[2][4] = {};
  int buf = 0;
  for (int k0 = 0; k0 < CC; k0 += BK, buf ^= 1) {
    if (k0 + BK < CC) {
      stage_tile(sA[buf ^ 1], Y, CC, mblk, k0 + BK);
      stage_tile(sB[buf ^ 1], W, CC, nblk, k0 + BK);
      __builtin_prefetch(W + (size_t)(nblk + (threadIdx.x >> 1)) * CC + k0 + 2 * BK, 0, 3);
    }
    v16bf a0 = fragA_lds(sA[buf], mloc);
    v16bf a1 = fragA_lds(sA[buf], mloc + 16);
#pragma unroll
    for (int j = 0; j < 4; ++j) {
      v16bf b = fragB_lds(sB[buf], nloc + j * 16);
      acc[0][j] = wmma_bf16(a0, b, acc[0][j]);
      acc[1][j] = wmma_bf16(a1, b, acc[1][j]);
    }
    stage_fence();
  }

  const int lane = threadIdx.x & 31;
  const int ncol = lane & 15;
  const int mr0 = (lane >> 4) * 8;
#pragma unroll
  for (int i = 0; i < 2; ++i)
#pragma unroll
    for (int j = 0; j < 4; ++j) {
      const int ng = nblk + nloc + j * 16 + ncol;
      const float bv = bias[ng];
#pragma unroll
      for (int r = 0; r < 8; ++r) {
        const int mg = mblk + mloc + i * 16 + mr0 + r;
        const size_t idx = (size_t)mg * CC + ng;
        const float val = acc[i][j][r] + bv + X[idx];
        X[idx] = val;
        Xb[idx] = f2bf(val);
      }
    }
}

// ----------------------------- host side -----------------------------------

extern "C" void kernel_launch(void* const* d_in, const int* in_sizes, int n_in,
                              void* d_out, int out_size, void* d_ws,
                              size_t ws_size, hipStream_t stream) {
  const float* x0    = (const float*)d_in[0];  // [4,2048,1024]
  const float* Wqkv  = (const float*)d_in[1];  // [12,3072,1024]
  const float* bqkv  = (const float*)d_in[2];  // [12,3072]
  const float* Wproj = (const float*)d_in[3];  // [12,1024,1024]
  const float* bproj = (const float*)d_in[4];  // [12,1024]
  float* out = (float*)d_out;                  // fp32 residual state + result

  char* ws = (char*)d_ws;
  size_t off = 0;
  auto wsalloc = [&](size_t bytes) -> void* {
    void* p = (void*)(ws + off);
    off += (bytes + 255) & ~(size_t)255;
    return p;
  };
  bf16_t* xb   = (bf16_t*)wsalloc((size_t)MTOK * CC * 2);          // 16 MiB
  bf16_t* yb   = (bf16_t*)wsalloc((size_t)MTOK * CC * 2);          // 16 MiB
  bf16_t* qb   = (bf16_t*)wsalloc((size_t)BB * HH * TT * DD * 2);  // 16 MiB
  bf16_t* kTb  = (bf16_t*)wsalloc((size_t)BB * HH * TT * DD * 2);  // 16 MiB
  bf16_t* vTb  = (bf16_t*)wsalloc((size_t)BB * HH * TT * DD * 2);  // 16 MiB
  bf16_t* sTb  = (bf16_t*)wsalloc((size_t)BB * HH * DD * DD * 2);  // 0.5 MiB
  bf16_t* wqkvb  = (bf16_t*)wsalloc((size_t)3 * CC * CC * 2);      // 6 MiB
  bf16_t* wprojb = (bf16_t*)wsalloc((size_t)CC * CC * 2);          // 2 MiB

  k_init<<<2048, 256, 0, stream>>>(x0, out, xb, MTOK * CC);

  const float scale = 0.125f;  // 1/sqrt(64)
  for (int l = 0; l < NLAYER; ++l) {
    k_cvt<<<2048, 256, 0, stream>>>(Wqkv + (size_t)l * 3 * CC * CC, wqkvb,
                                    3 * CC * CC);
    k_cvt<<<1024, 256, 0, stream>>>(Wproj + (size_t)l * CC * CC, wprojb,
                                    CC * CC);
    k_gemm_qkv<<<dim3(3 * CC / 128, MTOK / 128), 256, 0, stream>>>(
        xb, wqkvb, bqkv + (size_t)l * 3 * CC, qb, kTb, vTb);
    k_gemm_svt<<<BB * HH, 256, 0, stream>>>(vTb, kTb, sTb, scale);
    k_gemm_y<<<dim3(BB * HH, TT / 128), 256, 0, stream>>>(qb, sTb, yb);
    k_gemm_proj<<<dim3(CC / 128, MTOK / 128), 256, 0, stream>>>(
        yb, wprojb, bproj + (size_t)l * CC, out, xb);
  }
}
